// Pair_emb_w_templ_41025527611467
// MI455X (gfx1250) — matrix operand order repaired and checked by reading, not compiled, not run on hardware
//
#include <hip/hip_runtime.h>

// ---------------------------------------------------------------------------
// Pair embedding with template features, optimized for MI455X (gfx1250).
//   out[b,i,j,:] = rowvec[b,i,:] + colvec[b,j,:] + seqsep(b,i,j)*w_sep[:]
//                  + LN(templ[b,i,j,:]) @ (gamma .* W_T)
// rowvec folds: e@W_left + proj_b + beta@W_T + pos-enc(i) (ch 0..63)
// colvec folds: e@W_right + pos-enc(j) (ch 64..127)
// Main GEMM runs on v_wmma_f32_16x16x32_f16 (K=64 -> 2 WMMAs / 16x16 tile).
// Wave mapping: each wave32 owns one (b, j-tile) and sweeps 8 consecutive i,
// keeping colvec/w_sep/B-fragments register-resident across its 8 tiles.
// ---------------------------------------------------------------------------

typedef __attribute__((ext_vector_type(16))) _Float16 v16h;
typedef __attribute__((ext_vector_type(8)))  float    v8f;

#define BB 2
#define LL 512
#define DT 64      // d_templ
#define DM 128     // d_model
#define DE 64      // d_emb

union AFrag { v16h v; _Float16 h[16]; };
union Acc   { v8f  v; float    f[8];  };

// ---------------------------------------------------------------------------
// Kernel 1: per-(b,l) row/col vectors (128 ch each) with all rank-1 terms folded.
// ---------------------------------------------------------------------------
__global__ __launch_bounds__(128) void pair_pre(
    const int*   __restrict__ seq,
    const int*   __restrict__ idx,
    const float* __restrict__ emb_table,
    const float* __restrict__ ln_beta,
    const float* __restrict__ proj_w,   // [193,128] row-major
    const float* __restrict__ proj_b,
    float* __restrict__ rowvec,         // [B*L,128]
    float* __restrict__ colvec)         // [B*L,128]
{
  __shared__ float e[DE];
  const int bl = blockIdx.x;            // b*L + l
  const int c  = threadIdx.x;           // 0..127 channel
  if (c < DE) e[c] = emb_table[seq[bl] * DE + c];
  __syncthreads();

  float accR = proj_b[c];
  float accC = 0.0f;
#pragma unroll 8
  for (int k = 0; k < DE; ++k) {
    const float ek = e[k];
    accR += ek * proj_w[k * DM + c];          // left weights rows 0..63
    accC += ek * proj_w[(DE + k) * DM + c];   // right weights rows 64..127
  }
  // fold beta @ W_T (rows 129..192) into rowvec
  float accB = 0.0f;
#pragma unroll 8
  for (int k = 0; k < DT; ++k) accB += ln_beta[k] * proj_w[(129 + k) * DM + c];
  accR += accB;

  // 2D positional encoding terms (eval mode, dropout = identity)
  // pe(c') for c' in 0..63: m=c'&31, arg = idx * 10000^(-m/32); sin if c'<32 else cos
  const float x = (float)idx[bl];
  const int   m = c & 31;
  const float arg = x * expf(-(float)m * (9.210340371976184f / 32.0f));
  const float pe  = (c & 32) ? cosf(arg) : sinf(arg);

  rowvec[(size_t)bl * DM + c] = accR + ((c < 64)  ? pe : 0.0f);  // pe_left  indexed by i
  colvec[(size_t)bl * DM + c] = accC + ((c >= 64) ? pe : 0.0f);  // pe_right indexed by j
}

// ---------------------------------------------------------------------------
// Kernel 2: pre-pack gamma .* W_T into f16 WMMA B-fragments.
// Layout: wtx[((chunk*8 + n_tile)*32 + lane)*16 + t]
//   B 32x16 f16, contiguous k order used here (same as A): lanes 0-15 hold K=0..15 (2 per VGPR), lanes 16-31 K=16..31.
// ---------------------------------------------------------------------------
__global__ __launch_bounds__(256) void pair_wt(
    const float* __restrict__ proj_w,
    const float* __restrict__ ln_gamma,
    _Float16* __restrict__ wtx)
{
  const int u    = blockIdx.x * 256 + threadIdx.x;   // 0..8191
  const int t    = u & 15;
  const int lane = (u >> 4) & 31;
  const int n    = (u >> 9) & 7;
  const int c    = u >> 12;                          // K chunk (0 or 1)
  const int kB   = t + ((lane >> 4) ? 16 : 0);       // K within chunk
  const int K    = 32 * c + kB;                      // templ channel
  const int N    = 16 * n + (lane & 15);
  wtx[u] = (_Float16)(ln_gamma[K] * proj_w[(129 + K) * DM + N]);
}

// ---------------------------------------------------------------------------
// Kernel 3: main pair kernel. One wave32 per (b, j-tile), sweeping 8 i's.
// ---------------------------------------------------------------------------
struct WaveLds {
  float tile[16][68];   // 16 rows x 64 templ channels, padded stride 68
  float mean[16];
  float rstd[16];
  float ssep[16];
};

__global__ __launch_bounds__(256) void pair_main(
    const float*    __restrict__ templ,
    const int*      __restrict__ idx,
    const float*    __restrict__ proj_w,
    const _Float16* __restrict__ wtx,
    const float*    __restrict__ rowvec,
    const float*    __restrict__ colvec,
    float*          __restrict__ out)
{
  __shared__ WaveLds lds[8];
  const int lane   = threadIdx.x & 31;
  const int wave   = threadIdx.x >> 5;
  const int laneLo = lane & 15;
  const int laneHi = lane >> 4;
  WaveLds& W = lds[wave];

  const v16h* __restrict__ wtxv = (const v16h*)wtx;

  // 4096 waves total = 64 groups (2 b x 32 j-tiles) x 64 waves; each wave
  // covers 8 consecutive i rows. Launch must be 512 blocks x 256 threads.
  const int waveGlobal = blockIdx.x * 8 + wave;
  const int group = waveGlobal >> 6;       // 0..63 -> (b, j-tile)
  const int islot = waveGlobal & 63;       // 0..63 -> i block
  const int b  = group >> 5;
  const int j0 = (group & 31) * 16;
  const int i0 = islot * 8;

  // ---- j-fixed state, loaded once per wave ----
  // colvec 16x128 tile -> registers: colv[r][n] for M=r+8*laneHi, N=16n+laneLo
  float colv[8][8];
  {
    const float* cbase = colvec + (size_t)(b * LL + j0 + 8 * laneHi) * DM + laneLo;
#pragma unroll
    for (int r = 0; r < 8; ++r)
#pragma unroll
      for (int n = 0; n < 8; ++n)
        colv[r][n] = cbase[r * DM + 16 * n];
  }
  float ws[8];
#pragma unroll
  for (int n = 0; n < 8; ++n) ws[n] = proj_w[128 * DM + 16 * n + laneLo];

  // B fragments (gamma .* W_T), register-resident across all 8 tiles
  AFrag bf0[8], bf1[8];
#pragma unroll
  for (int n = 0; n < 8; ++n) {
    bf0[n].v = wtxv[(n    ) * 32 + lane];
    bf1[n].v = wtxv[(8 + n) * 32 + lane];
  }

  const int idxj = (laneHi == 0) ? idx[b * LL + j0 + laneLo] : 0;

  for (int it = 0; it < 8; ++it) {
    const int i = i0 + it;

    // --- seq-separation for the 16 j's of this tile (lanes 0-15) ---
    const int idx_i = idx[b * LL + i];      // wave-uniform scalar load
    if (laneHi == 0) {
      int d = idx_i - idxj;
      if (d < 0) d = -d;
      W.ssep[laneLo] = logf((float)d + 1.0f);
    }

    // --- coalesced load of the 16x64 templ tile into LDS (float4) ---
    const float* tbase = templ + (((size_t)(b * LL + i) * LL) + j0) * DT;
#pragma unroll
    for (int s = 0; s < 8; ++s) {
      const int q   = lane + 32 * s;
      const int row = q >> 4;
      const int c4  = (q & 15) * 4;
      const float4 v = *(const float4*)(tbase + (size_t)row * DT + c4);
      *(float4*)&W.tile[row][c4] = v;
    }

    // prefetch next i's templ tile (same j block, +L*DT floats)
    if (it < 7) __builtin_prefetch(tbase + (size_t)LL * DT + lane * 32, 0, 1);
    __syncthreads();

    // --- per-row LayerNorm statistics (lanes 0-15, one row each) ---
    if (laneHi == 0) {
      float s = 0.0f, s2 = 0.0f;
#pragma unroll 8
      for (int k = 0; k < DT; ++k) {
        const float x = W.tile[laneLo][k];
        s += x; s2 += x * x;
      }
      const float mu  = s * (1.0f / DT);
      const float var = s2 * (1.0f / DT) - mu * mu;
      W.mean[laneLo] = mu;
      W.rstd[laneLo] = rsqrtf(var + 1e-5f);
    }
    __syncthreads();

    // --- build A fragments (16x32 f16 A, contiguous k order matching wtx B) ---
    // lanes 0-15: row M=lane, K 0-15; lanes 16-31: K 16-31
    AFrag a0, a1;
    {
      const int   mrow = laneLo;
      const float mu = W.mean[mrow];
      const float rs = W.rstd[mrow];
#pragma unroll
      for (int tt = 0; tt < 16; ++tt) {
        const int kl = tt + (laneHi ? 16 : 0);
        a0.h[tt] = (_Float16)((W.tile[mrow][kl]      - mu) * rs);
        a1.h[tt] = (_Float16)((W.tile[mrow][kl + 32] - mu) * rs);
      }
    }

    // per-i vectors
    float ss[8];
#pragma unroll
    for (int r = 0; r < 8; ++r) ss[r] = W.ssep[r + 8 * laneHi];
    float rv[8];
    {
      const float* rowv = rowvec + (size_t)(b * LL + i) * DM + laneLo;
#pragma unroll
      for (int n = 0; n < 8; ++n) rv[n] = rowv[16 * n];
    }

    const size_t outbase = (((size_t)(b * LL + i)) * LL + j0) * DM;

#pragma unroll
    for (int n = 0; n < 8; ++n) {
      // seed accumulator with rank-1 terms (C operand of WMMA)
      Acc acc;
#pragma unroll
      for (int r = 0; r < 8; ++r)
        acc.f[r] = rv[n] + colv[r][n] + ss[r] * ws[n];

      acc.v = __builtin_amdgcn_wmma_f32_16x16x32_f16(
          false, a0.v, false, bf0[n].v, (short)0, acc.v, false, false);
      acc.v = __builtin_amdgcn_wmma_f32_16x16x32_f16(
          false, a1.v, false, bf1[n].v, (short)0, acc.v, false, false);

#pragma unroll
      for (int r = 0; r < 8; ++r)
        out[outbase + (size_t)(r + 8 * laneHi) * DM + 16 * n + laneLo] = acc.f[r];
    }
    __syncthreads();   // keep waves uniform before LDS reuse next iteration
  }
}

// ---------------------------------------------------------------------------
extern "C" void kernel_launch(void* const* d_in, const int* in_sizes, int n_in,
                              void* d_out, int out_size, void* d_ws, size_t ws_size,
                              hipStream_t stream) {
  (void)in_sizes; (void)n_in; (void)out_size; (void)ws_size;
  const int*   seq       = (const int*)d_in[0];
  const int*   idx       = (const int*)d_in[1];
  const float* templ     = (const float*)d_in[2];
  const float* emb_table = (const float*)d_in[3];
  const float* ln_gamma  = (const float*)d_in[4];
  const float* ln_beta   = (const float*)d_in[5];
  const float* proj_w    = (const float*)d_in[6];
  const float* proj_b    = (const float*)d_in[7];
  float* out = (float*)d_out;

  // workspace layout: [wtx f16: 8192 halves = 16KB][rowvec 512KB][colvec 512KB]
  _Float16* wtx    = (_Float16*)d_ws;
  float*    rowvec = (float*)((char*)d_ws + 8192 * sizeof(_Float16));
  float*    colvec = rowvec + (size_t)BB * LL * DM;

  pair_pre<<<BB * LL, 128, 0, stream>>>(seq, idx, emb_table, ln_beta, proj_w, proj_b,
                                        rowvec, colvec);
  pair_wt<<<32, 256, 0, stream>>>(proj_w, ln_gamma, wtx);
  pair_main<<<512, 256, 0, stream>>>(templ, idx, proj_w, wtx, rowvec, colvec, out);
}
